// ChordAwareAttention_45208825757800
// MI455X (gfx1250) — matrix operand order repaired and checked by reading, not compiled, not run on hardware
//
#include <hip/hip_runtime.h>

// ---------------------------------------------------------------------------
// ChordAwareAttention for MI455X (gfx1250): bf16 WMMA everywhere.
// B=4, L=1024, D=1024, H=16, HD=64. Fused score-K = 192 (pq|hq|vq)·(k|cf|bf).
// Round 4: explicit global address-space pointers (global_load_b128, LOADcnt
// only, no flat/DScnt coupling) + 2-stage modulo software pipeline.
// Requires K % 64 == 0 (K is 1024 or 192 here).
// ---------------------------------------------------------------------------

#define BB   4
#define LL   1024
#define DD   1024
#define HH   16
#define HD   64
#define KCD  192            // fused score K-dim: 3 * HD

typedef __bf16 bf16;
typedef __attribute__((ext_vector_type(4)))  float f32x4;
typedef __attribute__((ext_vector_type(8)))  float v8f;
typedef __attribute__((ext_vector_type(8)))  bf16  bf16x8;
typedef __attribute__((ext_vector_type(16))) bf16  v16bf;

#define GAS __attribute__((address_space(1)))
typedef GAS const float*  gfc;
typedef GAS const bf16*   gbc;
typedef GAS const f32x4*  gf4c;
typedef GAS const bf16x8* gb8c;
typedef GAS float*        gfm;
typedef GAS bf16*         gbm;

// generic -> global addrspace (flat address == global address on AMDGCN)
#define TO_G(T, p) ((T)(unsigned long long)(p))

union Frag { v16bf v; bf16x8 h[2]; };

__device__ __forceinline__ bf16x8 cvt8g(gfc p) {
    f32x4 a = *(gf4c)p;
    f32x4 b = *(gf4c)(p + 4);
    bf16x8 r;
    r[0]=(bf16)a[0]; r[1]=(bf16)a[1]; r[2]=(bf16)a[2]; r[3]=(bf16)a[3];
    r[4]=(bf16)b[0]; r[5]=(bf16)b[1]; r[6]=(bf16)b[2]; r[7]=(bf16)b[3];
    return r;
}

// ---------------------------------------------------------------------------
// Weight transpose + f32->bf16 (so WMMA B-fragments are contiguous K runs).
// ---------------------------------------------------------------------------
__global__ __launch_bounds__(256)
void transpose_cvt_kernel(const float* __restrict__ W, bf16* __restrict__ WT) {
    __shared__ float tile[32][33];
    int tx = threadIdx.x, ty = threadIdx.y;            // block (32, 8)
    int x  = blockIdx.x * 32 + tx;
    int y0 = blockIdx.y * 32;
    #pragma unroll
    for (int j = 0; j < 32; j += 8)
        tile[ty + j][tx] = W[(size_t)(y0 + ty + j) * DD + x];
    __syncthreads();
    int x2 = blockIdx.y * 32 + tx;
    int y2 = blockIdx.x * 32;
    #pragma unroll
    for (int j = 0; j < 32; j += 8)
        WT[(size_t)(y2 + ty + j) * DD + x2] = (bf16)tile[tx][ty + j];
}

// ---------------------------------------------------------------------------
// Chord / bass features (K=12, VALU): scatter into KC slots 1 and 2 for all b.
// ---------------------------------------------------------------------------
__global__ __launch_bounds__(256)
void chord_feats_kernel(const int* __restrict__ pc, const int* __restrict__ bo,
                        const float* __restrict__ Wc, const float* __restrict__ bc,
                        const float* __restrict__ Wb, const float* __restrict__ bb,
                        bf16* __restrict__ KC) {
    int idx  = blockIdx.x * 256 + threadIdx.x;         // L*D threads
    int dcol = idx & (DD - 1);
    int l    = idx >> 10;
    float accC = bc[dcol], accB = bb[dcol];
    #pragma unroll
    for (int j = 0; j < 12; j++) {
        accC += (float)pc[l * 12 + j] * Wc[j * DD + dcol];
        accB += (float)bo[l * 12 + j] * Wb[j * DD + dcol];
    }
    int h = dcol >> 6, w = dcol & 63;
    bf16 vc = (bf16)accC, vb = (bf16)accB;
    #pragma unroll
    for (int b = 0; b < BB; b++) {
        size_t base = ((size_t)(b * LL + l) * HH + h) * KCD;
        KC[base + 64  + w] = vc;   // harmony slot
        KC[base + 128 + w] = vb;   // bass slot
    }
}

// ---------------------------------------------------------------------------
// Unified bf16 WMMA GEMM, 2-stage modulo-scheduled software pipeline.
// Each wave: 32 rows x 64 cols = 2x4 tiles of v_wmma_f32_16x16x32_bf16.
// A row-major (f32 or bf16), B given transposed (N x K, bf16, contiguous K).
// All streaming accesses use explicit global-addrspace pointers so the
// backend emits global_load_b128 (LOADcnt only), never flat.
// Output MODEs:
//  0: grouped bf16  -> out[row*3072 + (col/64)*192 + slot*64 + col%64]  (QC/KC)
//  1: V-transposed  -> out[((b*H+h)*64+hd)*L + q]                       (VT)
//  2: plain f32+bias-> out[row*D + col]                                 (final)
//  3: scores f32    -> out[p*L*L + row*L + col] * (1/24)                (attn)
//  4: context bf16  -> out[(b*L+row)*D + h*64 + col]                    (CTX)
// Problem index p = blockIdx.z; bases = (p>>4)*_bs + (p&15)*_hs.
// ---------------------------------------------------------------------------
template<bool A_F32, int MODE, int WROWS, int WCOLS>
__global__ __launch_bounds__(32 * WROWS * WCOLS)
void gemm_wmma_kernel(const void* __restrict__ Ap, const bf16* __restrict__ BT,
                      const float* __restrict__ bias, void* __restrict__ outp,
                      int K, int lda, int ldbt,
                      long long a_bs, long long a_hs,
                      long long bt_bs, long long bt_hs, int slot) {
    int tid  = threadIdx.x;
    int wid  = tid >> 5;              // wave32
    int lane = tid & 31;
    int lr   = lane & 15;
    int sel  = lane >> 4;
    int wrow = wid / WCOLS, wcol = wid % WCOLS;
    int mbase = (blockIdx.y * WROWS + wrow) * 32;   // 2 M-tiles per wave
    int nbase = (blockIdx.x * WCOLS + wcol) * 64;   // 4 N-tiles per wave
    int p = blockIdx.z;
    long long abase  = (long long)(p >> 4) * a_bs  + (long long)(p & 15) * a_hs;
    long long btbase = (long long)(p >> 4) * bt_bs + (long long)(p & 15) * bt_hs;

    // Global-AS base pointers (advanced by 64 K-elements per outer iteration
    // so all loads inside the loop use literal instruction offsets).
    gfc paf[2];
    gbc pab[2];
    gbc pbp[4];
    #pragma unroll
    for (int m = 0; m < 2; m++) {
        long long ro = abase + (long long)(mbase + m * 16 + lr) * lda + sel * 8;
        paf[m] = TO_G(gfc, (const float*)Ap + ro);
        pab[m] = TO_G(gbc, (const bf16*)Ap + ro);
    }
    #pragma unroll
    for (int t = 0; t < 4; t++)
        pbp[t] = TO_G(gbc, BT + btbase +
                           (long long)(nbase + t * 16 + lr) * ldbt + sel * 8);

    v8f  acc[2][4] = {};
    Frag a0[2], a1[2], b0[4], b1[4];

    // kof must be a compile-time constant at each call site (fully unrolled).
    #define LOAD_A(dst, kof)                                                   \
        _Pragma("unroll")                                                      \
        for (int m = 0; m < 2; m++) {                                          \
            if (A_F32) {                                                       \
                dst[m].h[0] = cvt8g(paf[m] + (kof));                           \
                dst[m].h[1] = cvt8g(paf[m] + (kof) + 16);                      \
            } else {                                                           \
                dst[m].h[0] = *(gb8c)(pab[m] + (kof));                         \
                dst[m].h[1] = *(gb8c)(pab[m] + (kof) + 16);                    \
            }                                                                  \
        }
    #define LOAD_B(dst, kof)                                                   \
        _Pragma("unroll")                                                      \
        for (int t = 0; t < 4; t++) {                                          \
            dst[t].h[0] = *(gb8c)(pbp[t] + (kof));                             \
            dst[t].h[1] = *(gb8c)(pbp[t] + (kof) + 16);                        \
        }
    #define DO_WMMA(af, bf)                                                    \
        _Pragma("unroll")                                                      \
        for (int m = 0; m < 2; m++) {                                          \
            _Pragma("unroll")                                                  \
            for (int t = 0; t < 4; t++) {                                      \
                acc[m][t] = __builtin_amdgcn_wmma_f32_16x16x32_bf16(           \
                    false, af[m].v, false, bf[t].v,                            \
                    (short)0, acc[m][t], false, false);                        \
            }                                                                  \
        }

    // Prologue: fill both stages.
    LOAD_A(a0, 0)
    LOAD_B(b0, 0)
    LOAD_A(a1, 32)
    LOAD_B(b1, 32)
    for (int k0 = 0; k0 < K; k0 += 64) {
        bool more = (k0 + 64) < K;
        DO_WMMA(a0, b0)            // consumes loads issued one stage earlier
        if (more) {
            LOAD_A(a0, 64)         // refill just-consumed stage, 64 ahead
            LOAD_B(b0, 64)
        }
        DO_WMMA(a1, b1)
        if (more) {
            LOAD_A(a1, 96)
            LOAD_B(b1, 96)
        }
        #pragma unroll
        for (int m = 0; m < 2; m++) { paf[m] += 64; pab[m] += 64; }
        #pragma unroll
        for (int t = 0; t < 4; t++) pbp[t] += 64;
    }
    #undef LOAD_A
    #undef LOAD_B
    #undef DO_WMMA

    gbm outb = TO_G(gbm, (bf16*)outp);
    gfm outf = TO_G(gfm, (float*)outp);
    #pragma unroll
    for (int m = 0; m < 2; m++) {
        #pragma unroll
        for (int t = 0; t < 4; t++) {
            #pragma unroll
            for (int r = 0; r < 8; r++) {
                int row = mbase + m * 16 + r + sel * 8;
                int col = nbase + t * 16 + lr;
                float val = acc[m][t][r];
                if (bias) val += bias[col];
                if (MODE == 0) {
                    size_t idx = (size_t)row * (HH * KCD) + (col >> 6) * KCD +
                                 slot * 64 + (col & 63);
                    outb[idx] = (bf16)val;
                } else if (MODE == 1) {
                    int bb2 = row >> 10, q = row & (LL - 1);
                    int h = col >> 6,  hd = col & 63;
                    size_t idx = ((size_t)((bb2 * HH + h) * HD + hd)) * LL + q;
                    outb[idx] = (bf16)val;
                } else if (MODE == 2) {
                    outf[(size_t)row * DD + col] = val;
                } else if (MODE == 3) {
                    size_t idx = (size_t)p * LL * LL + (size_t)row * LL + col;
                    outf[idx] = val * (1.0f / 24.0f);  // (HD^-0.5)/3
                } else { // MODE == 4
                    int bb2 = p >> 4, h = p & 15;
                    size_t idx = ((size_t)(bb2 * LL + row)) * DD + h * 64 + col;
                    outb[idx] = (bf16)val;
                }
            }
        }
    }
}

// ---------------------------------------------------------------------------
// Row softmax over attn (in place), 1024 columns, one block per row.
// ---------------------------------------------------------------------------
__global__ __launch_bounds__(256)
void softmax_rows_kernel(float* __restrict__ attn) {
    __shared__ float red[256];
    gfm p = TO_G(gfm, attn + (size_t)blockIdx.x * LL);
    int t = threadIdx.x;
    f32x4 v = *(GAS const f32x4*)(p + t * 4);
    float m = fmaxf(fmaxf(v[0], v[1]), fmaxf(v[2], v[3]));
    red[t] = m; __syncthreads();
    for (int s = 128; s > 0; s >>= 1) {
        if (t < s) red[t] = fmaxf(red[t], red[t + s]);
        __syncthreads();
    }
    m = red[0]; __syncthreads();
    f32x4 e;
    float sum = 0.f;
    #pragma unroll
    for (int i = 0; i < 4; i++) { e[i] = __expf(v[i] - m); sum += e[i]; }
    red[t] = sum; __syncthreads();
    for (int s = 128; s > 0; s >>= 1) {
        if (t < s) red[t] += red[t + s];
        __syncthreads();
    }
    float inv = 1.0f / red[0];
    #pragma unroll
    for (int i = 0; i < 4; i++) e[i] *= inv;
    *(GAS f32x4*)(p + t * 4) = e;
}

// ---------------------------------------------------------------------------
extern "C" void kernel_launch(void* const* d_in, const int* in_sizes, int n_in,
                              void* d_out, int out_size, void* d_ws, size_t ws_size,
                              hipStream_t stream) {
    const float* query = (const float*)d_in[0];
    const float* key   = (const float*)d_in[1];
    const float* value = (const float*)d_in[2];
    const int*   pc    = (const int*)d_in[3];
    const int*   bon   = (const int*)d_in[4];
    const float* Wpq = (const float*)d_in[5];  const float* bpq = (const float*)d_in[6];
    const float* Whq = (const float*)d_in[7];  const float* bhq = (const float*)d_in[8];
    const float* Wvq = (const float*)d_in[9];  const float* bvq = (const float*)d_in[10];
    const float* Wk  = (const float*)d_in[11]; const float* bk  = (const float*)d_in[12];
    const float* Wv  = (const float*)d_in[13]; const float* bv  = (const float*)d_in[14];
    const float* Wc  = (const float*)d_in[15]; const float* bc  = (const float*)d_in[16];
    const float* Wb  = (const float*)d_in[17]; const float* bb  = (const float*)d_in[18];
    const float* Wo  = (const float*)d_in[19]; const float* bo  = (const float*)d_in[20];

    // workspace layout
    const size_t DD2 = (size_t)DD * DD * sizeof(bf16);       // 2 MB
    char* w = (char*)d_ws;
    bf16* WpqT = (bf16*)w; w += DD2;
    bf16* WhqT = (bf16*)w; w += DD2;
    bf16* WvqT = (bf16*)w; w += DD2;
    bf16* WkT  = (bf16*)w; w += DD2;
    bf16* WvT  = (bf16*)w; w += DD2;
    bf16* WoT  = (bf16*)w; w += DD2;
    bf16* QC   = (bf16*)w; w += (size_t)BB * LL * HH * KCD * sizeof(bf16);
    bf16* KC   = (bf16*)w; w += (size_t)BB * LL * HH * KCD * sizeof(bf16);
    bf16* VT   = (bf16*)w; w += (size_t)BB * HH * HD * LL * sizeof(bf16);
    bf16* CTX  = (bf16*)w; w += (size_t)BB * LL * DD * sizeof(bf16);

    float* out0  = (float*)d_out;
    float* attnp = out0 + (size_t)BB * LL * DD;

    // 1) transpose+convert all weights
    dim3 tb(32, 8), tg(DD / 32, DD / 32);
    transpose_cvt_kernel<<<tg, tb, 0, stream>>>(Wpq, WpqT);
    transpose_cvt_kernel<<<tg, tb, 0, stream>>>(Whq, WhqT);
    transpose_cvt_kernel<<<tg, tb, 0, stream>>>(Wvq, WvqT);
    transpose_cvt_kernel<<<tg, tb, 0, stream>>>(Wk,  WkT);
    transpose_cvt_kernel<<<tg, tb, 0, stream>>>(Wv,  WvT);
    transpose_cvt_kernel<<<tg, tb, 0, stream>>>(Wo,  WoT);

    // 2) chord / bass features into KC slots 1,2
    chord_feats_kernel<<<(LL * DD) / 256, 256, 0, stream>>>(pc, bon, Wc, bc, Wb, bb, KC);

    // 3) projections (A = f32 inputs, M=4096, N=1024, K=1024)
    //    block tile 64 x 256 (2x4 waves, each 32x64)
    dim3 pg(DD / 256, (BB * LL) / 64, 1);
    gemm_wmma_kernel<true, 0, 2, 4><<<pg, 256, 0, stream>>>(
        query, WpqT, bpq, QC, DD, DD, DD, 0, 0, 0, 0, 0);
    gemm_wmma_kernel<true, 0, 2, 4><<<pg, 256, 0, stream>>>(
        query, WhqT, bhq, QC, DD, DD, DD, 0, 0, 0, 0, 1);
    gemm_wmma_kernel<true, 0, 2, 4><<<pg, 256, 0, stream>>>(
        query, WvqT, bvq, QC, DD, DD, DD, 0, 0, 0, 0, 2);
    gemm_wmma_kernel<true, 0, 2, 4><<<pg, 256, 0, stream>>>(
        key,   WkT,  bk,  KC, DD, DD, DD, 0, 0, 0, 0, 0);
    gemm_wmma_kernel<true, 1, 2, 4><<<pg, 256, 0, stream>>>(
        value, WvT,  bv,  VT, DD, DD, DD, 0, 0, 0, 0, 0);

    // 4) fused scores: per (b,h) GEMM, M=N=1024, K=192 -> scaled f32 attn
    {
        long long qc_bs = (long long)LL * HH * KCD;   // per-batch stride
        long long qc_hs = KCD;                        // per-head stride
        dim3 sg(LL / 256, LL / 64, BB * HH);
        gemm_wmma_kernel<false, 3, 2, 4><<<sg, 256, 0, stream>>>(
            QC, KC, nullptr, attnp, KCD, HH * KCD, HH * KCD,
            qc_bs, qc_hs, qc_bs, qc_hs, 0);
    }

    // 5) softmax rows (in place on d_out attn region)
    softmax_rows_kernel<<<BB * HH * LL, 256, 0, stream>>>(attnp);

    // 6) context: per (b,h), attn(f32, LxL) x VT(bf16, 64xL)^T -> CTX bf16
    //    block tile 256 x 64 (8x1 waves, each 32x64)
    {
        long long at_bs = (long long)HH * LL * LL, at_hs = (long long)LL * LL;
        long long vt_bs = (long long)HH * HD * LL, vt_hs = (long long)HD * LL;
        dim3 cg(1, LL / 256, BB * HH);
        gemm_wmma_kernel<true, 4, 8, 1><<<cg, 256, 0, stream>>>(
            attnp, VT, nullptr, CTX, LL, LL, LL, at_bs, at_hs, vt_bs, vt_hs, 0);
    }

    // 7) output projection: CTX(bf16) x WoT + bo -> f32 d_out
    gemm_wmma_kernel<false, 2, 2, 4><<<pg, 256, 0, stream>>>(
        CTX, WoT, bo, out0, DD, DD, DD, 0, 0, 0, 0, 0);

    (void)in_sizes; (void)n_in; (void)out_size; (void)ws_size;
}
